// DomainAttention_80590766342265
// MI455X (gfx1250) — matrix operand order repaired and checked by reading, not compiled, not run on hardware
//
#include <hip/hip_runtime.h>

// Problem constants (match reference)
#define B_DIM  8192
#define D_DIM  128
#define NCAT   8192
#define S_DIM  4
#define CHUNK  2048            // NCAT / S_DIM
#define ROWS_PER_WAVE 4096     // NCAT / 2 waves per block

typedef __attribute__((ext_vector_type(16))) __bf16 v16bf;
typedef __attribute__((ext_vector_type(8)))  float  v8f;

// WMMA fragment viewed either as bf16 vector (for the intrinsic) or as two
// 128-bit chunks (for raw global_load_b128 of pre-converted data).
union Frag {
  v16bf b;
  uint4 q[2];
};

// float -> bf16 (round to nearest even)
__device__ __forceinline__ unsigned int f2bf(float f) {
  unsigned int x = __float_as_uint(f);
  return (x + 0x7FFFu + ((x >> 16) & 1u)) >> 16;
}

// ---------------------------------------------------------------------------
// One-time conversion: f32 tensor -> bf16 (8 elements per thread)
// ---------------------------------------------------------------------------
__global__ __launch_bounds__(256)
void cvt_bf16_kernel(const float* __restrict__ src,
                     unsigned short* __restrict__ dst, int n8) {
  int i = blockIdx.x * blockDim.x + threadIdx.x;
  if (i >= n8) return;
  const float4* s = (const float4*)src + (size_t)i * 2;
  float4 f0 = s[0];
  float4 f1 = s[1];
  uint4 o;
  o.x = f2bf(f0.x) | (f2bf(f0.y) << 16);
  o.y = f2bf(f0.z) | (f2bf(f0.w) << 16);
  o.z = f2bf(f1.x) | (f2bf(f1.y) << 16);
  o.w = f2bf(f1.z) | (f2bf(f1.w) << 16);
  ((uint4*)dst)[i] = o;
}

// ---------------------------------------------------------------------------
// A-tile loader: 16 rows x 128 K of bf16 cat as 4 K=32 fragments (8 x b128).
// A (16x32 bf16): lane m = lane&15,
//   e<8  -> K = ks*32 + hh*8 + e
//   e>=8 -> K = ks*32 + 16 + hh*8 + (e-8)
// ---------------------------------------------------------------------------
__device__ __forceinline__ void load_a(const unsigned short* __restrict__ catb,
                                       int mbase, int lo, int hh, Frag a[4]) {
  const unsigned short* ap = catb + (size_t)(mbase + lo) * D_DIM;
#pragma unroll
  for (int ks = 0; ks < 4; ++ks) {
    const unsigned short* p = ap + ks * 32 + hh * 8;
    a[ks].q[0] = *(const uint4*)(p);        // K = base .. base+7
    a[ks].q[1] = *(const uint4*)(p + 16);   // K = base+16 .. base+23
  }
}

// 8 chained WMMAs: one 16(M) x 32-col tile of con = cat @ batch_x^T
__device__ __forceinline__ void wmma_tile(const Frag a[4], const Frag bfr[2][4],
                                          v8f acc[2]) {
  v8f zero = {0.f, 0.f, 0.f, 0.f, 0.f, 0.f, 0.f, 0.f};
  acc[0] = zero;
  acc[1] = zero;
#pragma unroll
  for (int ks = 0; ks < 4; ++ks) {
    acc[0] = __builtin_amdgcn_wmma_f32_16x16x32_bf16(
        false, a[ks].b, false, bfr[0][ks].b, (short)0, acc[0], false, false);
    acc[1] = __builtin_amdgcn_wmma_f32_16x16x32_bf16(
        false, a[ks].b, false, bfr[1][ks].b, (short)0, acc[1], false, false);
  }
}

// Pass-1 per-tile reduction: sum(v^4) and max(v) per column
__device__ __forceinline__ void p1_reduce(const v8f acc[2], float s4[2],
                                          float mx[2]) {
#pragma unroll
  for (int t = 0; t < 2; ++t)
#pragma unroll
    for (int j = 0; j < 8; ++j) {
      float v  = acc[t][j];
      float v2 = v * v;
      s4[t] += v2 * v2;
      mx[t]  = fmaxf(mx[t], v);
    }
}

// Pass-2 per-tile reduction: softmax numerator terms
__device__ __forceinline__ void p2_reduce(const v8f acc[2],
                                          const float mc[2],
                                          const float invden[2],
                                          const float* __restrict__ yp,
                                          float Z[2], float* ws0, float* ws1) {
  float4 ya = *(const float4*)(yp);
  float4 yb = *(const float4*)(yp + 4);
  float yv[8] = {ya.x, ya.y, ya.z, ya.w, yb.x, yb.y, yb.z, yb.w};
#pragma unroll
  for (int j = 0; j < 8; ++j) {
    float p0 = __expf((acc[0][j] - mc[0]) * invden[0]);
    float p1 = __expf((acc[1][j] - mc[1]) * invden[1]);
    Z[0] += p0;
    Z[1] += p1;
    *ws0 += yv[j] * p0;
    *ws1 += yv[j] * p1;
  }
}

// ---------------------------------------------------------------------------
// Main fused kernel. Block = 64 threads = 2 waves; block owns 32 columns;
// wave w streams rows [w*4096, (w+1)*4096); reductions combined via LDS.
// A-tiles are double-buffered so loads of tile i+1 overlap WMMAs of tile i.
// NOTE: the pipeline intentionally over-reads one 16-row tile past each row
// range; those addresses stay inside the 4 MB bf16 workspace (catb+bxb).
// ---------------------------------------------------------------------------
__global__ __launch_bounds__(64)
void DomainAttention_kernel(const unsigned short* __restrict__ catb, // [NCAT,D] bf16
                            const unsigned short* __restrict__ bxb,  // [B,D] bf16
                            const float* __restrict__ bx,            // [B,D] f32
                            const float* __restrict__ y,             // [NCAT]
                            const float* __restrict__ phi,           // [S,D]
                            const float* __restrict__ bias,          // [1]
                            float* __restrict__ out) {               // [B]
  const int lane = threadIdx.x & 31;   // wave32
  const int wave = threadIdx.x >> 5;   // 0 or 1
  const int hh   = lane >> 4;          // lane half
  const int lo   = lane & 15;
  const int colbase = blockIdx.x * 32;
  const int rowbase = wave * ROWS_PER_WAVE;
  const int rowend  = rowbase + ROWS_PER_WAVE;

  __shared__ float ls4[2][2][16];
  __shared__ float lmx[2][2][16];
  __shared__ float lZ [2][2][16];
  __shared__ float lw [2][2][S_DIM][16];

  // ---- B fragments (batch_x columns, bf16), reused for all tiles/passes.
  // B (32x16): n = lane&15, e<8 -> K = ks*32 + hh*8 + e, e>=8 -> K = ks*32 + 16 + hh*8 + (e-8)
  Frag bfr[2][4];
#pragma unroll
  for (int t = 0; t < 2; ++t) {
    const unsigned short* bp = bxb + (size_t)(colbase + t * 16 + lo) * D_DIM;
#pragma unroll
    for (int ks = 0; ks < 4; ++ks) {
      const unsigned short* p = bp + ks * 32 + hh * 8;
      bfr[t][ks].q[0] = *(const uint4*)(p);
      bfr[t][ks].q[1] = *(const uint4*)(p + 16);
    }
  }

  // ---- Pass 1: per-column sum(|con|^4) and max(con) over this wave's rows.
  float s4[2] = {0.f, 0.f};
  float mx[2] = {-3.402823466e38f, -3.402823466e38f};
  {
    Frag a0[4], a1[4];
    load_a(catb, rowbase, lo, hh, a0);
    for (int mb = rowbase; mb < rowend; mb += 32) {
      load_a(catb, mb + 16, lo, hh, a1);       // overlap with tile mb
      {
        v8f acc[2];
        wmma_tile(a0, bfr, acc);
        p1_reduce(acc, s4, mx);
      }
      load_a(catb, mb + 32, lo, hh, a0);       // overlap with tile mb+16
      {
        v8f acc[2];
        wmma_tile(a1, bfr, acc);
        p1_reduce(acc, s4, mx);
      }
    }
  }
  // intra-wave (cross-half) then cross-wave (LDS) combine
#pragma unroll
  for (int t = 0; t < 2; ++t) {
    s4[t] += __shfl_xor(s4[t], 16, 32);
    mx[t]  = fmaxf(mx[t], __shfl_xor(mx[t], 16, 32));
  }
  if (hh == 0) {
#pragma unroll
    for (int t = 0; t < 2; ++t) {
      ls4[wave][t][lo] = s4[t];
      lmx[wave][t][lo] = mx[t];
    }
  }
  __syncthreads();
  float invden[2], mc[2];
#pragma unroll
  for (int t = 0; t < 2; ++t) {
    float s4t = ls4[0][t][lo] + ls4[1][t][lo];
    float mxt = fmaxf(lmx[0][t][lo], lmx[1][t][lo]);
    float n4  = sqrtf(sqrtf(s4t));      // ||col||_4
    float den = fmaxf(n4, 1e-12f);      // F.normalize eps clamp
    invden[t] = 1.0f / den;
    mc[t]     = mxt;                    // softmax stabilizer
  }

  // ---- Pass 2: softmax denominator Z, per-chunk y-weighted numerators.
  // exp(con/den - max/den) = exp((con - max) * invden)
  float Z[2] = {0.f, 0.f};
  float w[2][S_DIM];
#pragma unroll
  for (int s = 0; s < S_DIM; ++s) { w[0][s] = 0.f; w[1][s] = 0.f; }

#pragma unroll
  for (int s = 0; s < S_DIM; ++s) {
    // intersect this wave's row range with chunk s (bounds are multiples of 2048)
    int lob = s * CHUNK;
    int beg = lob > rowbase ? lob : rowbase;
    int end = (lob + CHUNK) < rowend ? (lob + CHUNK) : rowend;
    if (beg >= end) continue;
    float ws0 = 0.f, ws1 = 0.f;
    Frag a0[4], a1[4];
    load_a(catb, beg, lo, hh, a0);
    for (int mb = beg; mb < end; mb += 32) {
      load_a(catb, mb + 16, lo, hh, a1);
      {
        v8f acc[2];
        wmma_tile(a0, bfr, acc);
        p2_reduce(acc, mc, invden, y + mb + hh * 8, Z, &ws0, &ws1);
      }
      load_a(catb, mb + 32, lo, hh, a0);
      {
        v8f acc[2];
        wmma_tile(a1, bfr, acc);
        p2_reduce(acc, mc, invden, y + mb + 16 + hh * 8, Z, &ws0, &ws1);
      }
    }
    w[0][s] += ws0;
    w[1][s] += ws1;
  }
#pragma unroll
  for (int t = 0; t < 2; ++t) {
    Z[t] += __shfl_xor(Z[t], 16, 32);
#pragma unroll
    for (int s = 0; s < S_DIM; ++s)
      w[t][s] += __shfl_xor(w[t][s], 16, 32);
  }
  if (hh == 0) {
#pragma unroll
    for (int t = 0; t < 2; ++t) {
      lZ[wave][t][lo] = Z[t];
#pragma unroll
      for (int s = 0; s < S_DIM; ++s)
        lw[wave][t][s][lo] = w[t][s];
    }
  }
  __syncthreads();

  // ---- Epilogue: wave t handles N-tile t.
  // theta[s,b] = exp(bx[b].phi[s]);  y_hat = sum_s theta*w / Z + bias
  {
    const int t   = wave;
    const int col = colbase + t * 16 + lo;
    float Zt = lZ[0][t][lo] + lZ[1][t][lo];
    const float* bp = bx + (size_t)col * D_DIM;
    float accv = 0.f;
#pragma unroll
    for (int s = 0; s < S_DIM; ++s) {
      float wt = lw[0][t][s][lo] + lw[1][t][s][lo];
      const float* pp = phi + s * D_DIM;
      float dot = 0.f;
      for (int k = 0; k < D_DIM; k += 4) {
        float4 a = *(const float4*)(bp + k);
        float4 c = *(const float4*)(pp + k);
        dot += a.x * c.x + a.y * c.y + a.z * c.z + a.w * c.w;
      }
      accv += __expf(dot) * wt;
    }
    if (hh == 0) out[col] = accv / Zt + bias[0];
  }
}

// ---------------------------------------------------------------------------
extern "C" void kernel_launch(void* const* d_in, const int* in_sizes, int n_in,
                              void* d_out, int out_size, void* d_ws, size_t ws_size,
                              hipStream_t stream) {
  (void)in_sizes; (void)n_in; (void)out_size; (void)ws_size;
  const float* bx   = (const float*)d_in[0];  // batch_x [B, D]
  const float* cat  = (const float*)d_in[1];  // cat     [NCAT, D]
  const float* y    = (const float*)d_in[2];  // y       [NCAT]
  const float* phi  = (const float*)d_in[3];  // phi     [S, D]
  const float* bias = (const float*)d_in[4];  // bias    [1]
  float* out = (float*)d_out;                 // y_hat   [B]

  // Workspace layout: bf16 copies of cat and batch_x (2 MB each, contiguous;
  // the main kernel's pipeline over-read stays inside this 4 MB region)
  unsigned short* catb = (unsigned short*)d_ws;
  unsigned short* bxb  = catb + (size_t)NCAT * D_DIM;

  const int n8 = (NCAT * D_DIM) / 8;          // 131072 (same for both tensors)
  cvt_bf16_kernel<<<(n8 + 255) / 256, 256, 0, stream>>>(cat, catb, n8);
  cvt_bf16_kernel<<<(n8 + 255) / 256, 256, 0, stream>>>(bx, bxb, n8);

  dim3 block(64);                 // 2 waves; block owns 32 columns, rows split
  dim3 grid(B_DIM / 32);          // 256 blocks
  DomainAttention_kernel<<<grid, block, 0, stream>>>(catb, bxb, bx, y, phi,
                                                     bias, out);
}
